// LunarisMind_7584912244847
// MI455X (gfx1250) — matrix operand order repaired and checked by reading, not compiled, not run on hardware
//
#include <hip/hip_runtime.h>
#include <stddef.h>
#include <stdint.h>

// ---------------- model constants ----------------
static constexpr int SS   = 1024;
static constexpr int DD   = 1024;
static constexpr int HH   = 16;
static constexpr int HD   = 64;       // DD/HH
static constexpr int LL   = 10;
static constexpr int VV   = 32000;
static constexpr int RR   = 32;
static constexpr int DFF  = 4096;     // 4*DD
static constexpr int SPAD = SS + 16;  // padded key dim for V^T
static constexpr int VROWS = HD + 16; // V^T rows per head (+ones row for rowsum)
static constexpr float LORA_SCALE = 1.0f / 32.0f;

// ---------------- vector types ----------------
typedef __attribute__((ext_vector_type(16))) __bf16 v16bf;
typedef __attribute__((ext_vector_type(8)))  __bf16 v8bf;
typedef __attribute__((ext_vector_type(8)))  float  v8f;
typedef __attribute__((ext_vector_type(4)))  float  v4f;
typedef unsigned int u32;
typedef __attribute__((ext_vector_type(4))) u32 u32x4;
typedef __attribute__((ext_vector_type(8))) int i32x8;
typedef __attribute__((ext_vector_type(4))) int i32x4;

// ---------------- WMMA fragment loaders ----------------
// A-matrix 16x32 bf16 layout (ISA 7.12.2): lane<16 holds K {0..7,16..23},
// lane>=16 holds K {8..15,24..31}, for row M = lane&15.
__device__ __forceinline__ v16bf frag_a(const __bf16* __restrict__ p, int sel) {
  v8bf lo = *reinterpret_cast<const v8bf*>(p + sel * 8);
  v8bf hi = *reinterpret_cast<const v8bf*>(p + 16 + sel * 8);
  return __builtin_shufflevector(lo, hi, 0,1,2,3,4,5,6,7,8,9,10,11,12,13,14,15);
}

// B-matrix 32x16 bf16 layout: lane holds column N = lane&15,
// K {sel*8..sel*8+7, 16+sel*8..16+sel*8+7} (same order as A).
__device__ __forceinline__ v16bf load_bfrag(const __bf16* __restrict__ p, int sel) {
  v8bf lo = *reinterpret_cast<const v8bf*>(p + sel * 8);
  v8bf hi = *reinterpret_cast<const v8bf*>(p + 16 + sel * 8);
  return __builtin_shufflevector(lo, hi, 0,1,2,3,4,5,6,7,8,9,10,11,12,13,14,15);
}

__device__ __forceinline__ v8bf cvt8(const float* __restrict__ p) {
  v4f a = *reinterpret_cast<const v4f*>(p);
  v4f b = *reinterpret_cast<const v4f*>(p + 4);
  v8f f = __builtin_shufflevector(a, b, 0,1,2,3,4,5,6,7);
  return __builtin_convertvector(f, v8bf);
}

__device__ __forceinline__ v8f wmma_bf16(v16bf a, v16bf b, v8f c) {
  return __builtin_amdgcn_wmma_f32_16x16x32_bf16(false, a, false, b, (short)0, c,
                                                 false, false);
}

// ---------------- TDM: async tensor load of a 64x64 f32 tile into LDS ----------
// Builds D# groups per ISA ch.8: group0 {count=1, lds_addr, global_addr, type=2},
// group1 {data_size=4B, tensor dims/strides, tile 64(w) x 64(h)}.
// This toolchain exposes the 6-arg builtin form (g0, g1, g2, g3, g4, cpol).
__device__ __forceinline__ void tdm_load_tile(u32 lds_off,
                                              const float* __restrict__ gptr,
                                              int rows, int rowlen) {
  const unsigned long long ga = (unsigned long long)(uintptr_t)gptr;
  u32x4 g0;
  g0.x = 1u;                                            // count=1 (user D#)
  g0.y = lds_off;                                       // LDS byte address
  g0.z = (u32)(ga & 0xFFFFFFFFu);                       // global_addr[31:0]
  g0.w = (u32)((ga >> 32) & 0x1FFFFFFu) | (2u << 30);   // addr[56:32] | type=2
  u32 w0 = (2u << 16);                                  // data_size = 4 bytes
  const u32 td0 = (u32)rowlen;                          // tensor_dim0
  const u32 td1 = (u32)rows;                            // tensor_dim1
  u32 w1 = (td0 & 0xFFFFu) << 16;                       // bits 63:48
  u32 w2 = ((td0 >> 16) & 0xFFFFu) | ((td1 & 0xFFFFu) << 16);
  u32 w3 = ((td1 >> 16) & 0xFFFFu) | (64u << 16);       // tile_dim0 = 64
  u32 w4 = 64u;                                         // tile_dim1 = 64, dim2=0
  u32 w5 = (u32)rowlen;                                 // tensor_dim0_stride lo
  i32x8 g1 = {(int)w0, (int)w1, (int)w2, (int)w3,
              (int)w4, (int)w5, 0, 0};
  i32x4 z4 = {0, 0, 0, 0};
  i32x8 z8 = {0, 0, 0, 0, 0, 0, 0, 0};
  __builtin_amdgcn_tensor_load_to_lds(g0, g1, z4, z4, z8, 0);
}

// ---------------- TDM-staged NT GEMM: C[M,N] = A[M,K] * B[N,K]^T ----------------
// B is f32 weights. Wave 0 DMA-stages the block's shared 64-row B tile into a
// double-buffered f32 LDS tile via the Tensor Data Mover; all 256 threads then
// convert it ONCE to a bf16 LDS tile (instead of per-fragment cvt chains), and
// the WMMA loop streams pure bf16 fragments from LDS while the next DMA is in
// flight. Requires K%64==0, M%128==0, N%64==0 (true for every f32 GEMM here).
__global__ __launch_bounds__(256)
void k_gemm_tdm(const __bf16* __restrict__ A, int lda,
                const float* __restrict__ B, int ldb,
                float* __restrict__ C, int ldc, int M, int N, int K) {
  __shared__ float  Bf[2][64 * 64];   // 32 KB TDM double buffer (f32)
  __shared__ __bf16 Bb[64 * 64];      // 8 KB converted tile (bf16)
  const int lane = threadIdx.x & 31;
  const int wave = threadIdx.x >> 5;
  const int m0   = blockIdx.y * 128 + wave * 16;
  const int n0   = blockIdx.x * 64;
  const int row  = lane & 15;
  const int sel  = lane >> 4;
  const __bf16* pa = A + (size_t)(m0 + row) * lda;

  const int nchunk = K / 64;
  if (wave == 0)
    tdm_load_tile((u32)(uintptr_t)&Bf[0][0], B + (size_t)n0 * ldb, N - n0, ldb);

  v8f acc[4] = {};
  for (int c = 0; c < nchunk; ++c) {
    if (wave == 0) __builtin_amdgcn_s_wait_tensorcnt(0);  // tile c landed
    __syncthreads();                                      // (also: Bb free)

    // cooperative one-shot f32 -> bf16 conversion (16 elems per thread)
    {
      const float* s = &Bf[c & 1][0] + threadIdx.x * 16;
      v8bf* d = reinterpret_cast<v8bf*>(&Bb[0] + threadIdx.x * 16);
      d[0] = cvt8(s);
      d[1] = cvt8(s + 8);
    }
    __syncthreads();                                      // bf16 tile ready

    // overlap: next DMA in flight while we do WMMA on this tile
    if (wave == 0 && c + 1 < nchunk)
      tdm_load_tile((u32)(uintptr_t)&Bf[(c + 1) & 1][0],
                    B + (size_t)n0 * ldb + (c + 1) * 64, N - n0, ldb);

    const int kg = c * 64;
    if (kg + 64 < K) __builtin_prefetch(pa + kg + 64, 0, 1);
#pragma unroll
    for (int kk = 0; kk < 64; kk += 32) {
      v16bf a = frag_a(pa + kg + kk, sel);
#pragma unroll
      for (int j = 0; j < 4; ++j) {
        const __bf16* pb = &Bb[0] + (size_t)(16 * j + row) * 64 + kk;
        acc[j] = wmma_bf16(a, load_bfrag(pb, sel), acc[j]);
      }
    }
  }

  const int mbase = m0 + sel * 8;
#pragma unroll
  for (int j = 0; j < 4; ++j) {
    const int n = n0 + 16 * j + row;
    float* pc = C + (size_t)mbase * ldc + n;
#pragma unroll
    for (int r = 0; r < 8; ++r) pc[(size_t)r * ldc] = acc[j][r];
  }
}

// ---------------- generic NT GEMM (bf16 B) for the tiny LoRA mats ------------
template <bool ACC>
__global__ __launch_bounds__(256)
void k_gemm_nt(const __bf16* __restrict__ A, int lda,
               const __bf16* __restrict__ B, int ldb,
               float* __restrict__ C, int ldc, int M, int N, int K) {
  const int lane = threadIdx.x & 31;
  const int wave = threadIdx.x >> 5;
  const int m0   = blockIdx.y * 128 + wave * 16;
  const int n0   = blockIdx.x * 64;
  if (m0 >= M) return;
  const int row = lane & 15;
  const int sel = lane >> 4;
  const __bf16* pa = A + (size_t)(m0 + row) * lda;
  const int rem = N - n0;
  const int nt  = rem >= 64 ? 4 : ((rem + 15) >> 4);

  v8f acc[4] = {};
  for (int k0 = 0; k0 < K; k0 += 32) {
    v16bf a = frag_a(pa + k0, sel);
#pragma unroll
    for (int j = 0; j < 4; ++j) {
      if (j < nt) {
        const __bf16* pb = B + (size_t)(n0 + 16 * j + row) * ldb + k0;
        acc[j] = wmma_bf16(a, load_bfrag(pb, sel), acc[j]);
      }
    }
  }
  const int mbase = m0 + sel * 8;
#pragma unroll
  for (int j = 0; j < 4; ++j) {
    if (j < nt) {
      const int n = n0 + 16 * j + row;
      float* pc = C + (size_t)mbase * ldc + n;
#pragma unroll
      for (int r = 0; r < 8; ++r) {
        float v = acc[j][r];
        if (ACC) pc[(size_t)r * ldc] += v;
        else     pc[(size_t)r * ldc]  = v;
      }
    }
  }
}

// ---------------- fused flash attention (causal + ALiBi) ----------------
// One wave per 16-query tile per head; key tiles streamed in pairs (K=32 for
// the P*V WMMA). Online softmax in registers; P staged C->A layout via LDS.
// The softmax denominator rides for free in a 5th WMMA accumulator against an
// all-ones row appended to V^T (rowsum == column 0 of that tile).
__global__ __launch_bounds__(128)
void k_flash(const __bf16* __restrict__ Q, const __bf16* __restrict__ Kh,
             const __bf16* __restrict__ Vt, __bf16* __restrict__ O, int S) {
  __shared__ __bf16 Pls[4][16 * 32];
  const int lane = threadIdx.x & 31;
  const int wave = threadIdx.x >> 5;
  const int z    = blockIdx.y;                    // head
  const int i0   = (blockIdx.x * 4 + wave) * 16;  // query tile base
  const int row  = lane & 15;
  const int sel  = lane >> 4;
  const float slope = exp2f(-0.5f * (float)(z + 1));

  const __bf16* qa = Q + ((size_t)z * S + i0 + row) * HD;
  const v16bf qf0 = frag_a(qa, sel);
  const v16bf qf1 = frag_a(qa + 32, sel);

  float mrun[8];
  v8f oacc[5] = {};   // [0..3] = output d-tiles, [4] col0 = softmax denominator
#pragma unroll
  for (int r = 0; r < 8; ++r) mrun[r] = -3.0e38f;

  __bf16* pls = &Pls[wave][0];

  for (int j0 = 0; j0 <= i0; j0 += 32) {
    const bool hasB = (j0 + 16 <= i0);  // wave-uniform -> EXEC stays all-ones

    // ---- S-tiles = Q K^T ----
    const __bf16* ka = Kh + ((size_t)z * S + j0 + row) * HD;
    v8f sA = {};
    sA = wmma_bf16(qf0, load_bfrag(ka, sel), sA);
    sA = wmma_bf16(qf1, load_bfrag(ka + 32, sel), sA);
    v8f sB = {};
    if (hasB) {
      const __bf16* kb2 = ka + (size_t)16 * HD;
      sB = wmma_bf16(qf0, load_bfrag(kb2, sel), sB);
      sB = wmma_bf16(qf1, load_bfrag(kb2 + 32, sel), sB);
    }

    // ---- online softmax max-update (row groups live in 16-lane halves) ----
    const int nA = j0 + row;
    const int nB = j0 + 16 + row;
    float scl[8];
#pragma unroll
    for (int r = 0; r < 8; ++r) {
      const int m = i0 + sel * 8 + r;
      const float a = (nA <= m) ? sA[r] + slope * (float)(nA - m) : -3.0e38f;
      const float b = (hasB && nB <= m) ? sB[r] + slope * (float)(nB - m)
                                        : -3.0e38f;
      float mx = fmaxf(a, b);
      mx = fmaxf(mx, __shfl_xor(mx, 1, 32));
      mx = fmaxf(mx, __shfl_xor(mx, 2, 32));
      mx = fmaxf(mx, __shfl_xor(mx, 4, 32));
      mx = fmaxf(mx, __shfl_xor(mx, 8, 32));
      const float mnew = fmaxf(mrun[r], mx);
      scl[r] = __expf(mrun[r] - mnew);
      mrun[r] = mnew;
      const float ea = (a > -1.0e38f) ? __expf(a - mnew) : 0.0f;
      const float eb = (b > -1.0e38f) ? __expf(b - mnew) : 0.0f;
      // stage P row-major [16][32] in LDS (second half zero on diagonal tail)
      pls[(sel * 8 + r) * 32 + row]      = (__bf16)ea;
      pls[(sel * 8 + r) * 32 + 16 + row] = (__bf16)eb;
    }

    // rescale running output + denominator
#pragma unroll
    for (int t = 0; t < 5; ++t)
#pragma unroll
      for (int r = 0; r < 8; ++r) oacc[t][r] *= scl[r];

    // ---- O += P V (K = 32 keys); tile 4 hits the ones row -> rowsum ----
    const v16bf pf = frag_a(pls + row * 32, sel);  // same-wave LDS RAW: in-order
#pragma unroll
    for (int t = 0; t < 5; ++t) {
      const __bf16* vb = Vt + ((size_t)z * VROWS + t * 16 + row) * SPAD + j0;
      oacc[t] = wmma_bf16(pf, load_bfrag(vb, sel), oacc[t]);
    }
  }

  // denominator: column 0 of tile 4 lives in lane sel*16; broadcast per group
  float lsum[8];
#pragma unroll
  for (int r = 0; r < 8; ++r) lsum[r] = __shfl(oacc[4][r], sel * 16, 32);

  // ---- normalize and store bf16, heads merged into [S, D] ----
#pragma unroll
  for (int t = 0; t < 4; ++t) {
#pragma unroll
    for (int r = 0; r < 8; ++r) {
      const int m = i0 + sel * 8 + r;
      O[(size_t)m * DD + z * HD + t * 16 + row] =
          (__bf16)(oacc[t][r] / lsum[r]);
    }
  }
}

// ---------------- LayerNorm -> bf16 ----------------
__global__ __launch_bounds__(256)
void k_layernorm(const float* __restrict__ x, const float* __restrict__ g,
                 const float* __restrict__ b, __bf16* __restrict__ out, int D) {
  __shared__ float red[256];
  const int row = blockIdx.x;
  const int tid = threadIdx.x;
  const float* xr = x + (size_t)row * D;

  float s = 0.f;
  for (int i = tid; i < D; i += 256) s += xr[i];
  red[tid] = s; __syncthreads();
  for (int w = 128; w > 0; w >>= 1) {
    if (tid < w) red[tid] += red[tid + w];
    __syncthreads();
  }
  const float mean = red[0] / (float)D; __syncthreads();

  float ss = 0.f;
  for (int i = tid; i < D; i += 256) { float d = xr[i] - mean; ss += d * d; }
  red[tid] = ss; __syncthreads();
  for (int w = 128; w > 0; w >>= 1) {
    if (tid < w) red[tid] += red[tid + w];
    __syncthreads();
  }
  const float r = rsqrtf(red[0] / (float)D + 1e-6f);

  __bf16* o = out + (size_t)row * D;
  for (int i = tid; i < D; i += 256)
    o[i] = (__bf16)((xr[i] - mean) * r * g[i] + b[i]);
}

// ---------------- small elementwise kernels ----------------
__global__ void k_embed(const int* __restrict__ ids, const float* __restrict__ emb,
                        float* __restrict__ x, int S, int D) {
  const int idx = blockIdx.x * 256 + threadIdx.x;
  if (idx >= S * D) return;
  const int s = idx / D, d = idx - s * D;
  x[idx] = emb[(size_t)ids[s] * D + d];
}

__global__ void k_cvt(const float* __restrict__ in, __bf16* __restrict__ out, int n) {
  const int idx = blockIdx.x * 256 + threadIdx.x;
  if (idx < n) out[idx] = (__bf16)in[idx];
}

// out[c*rows + r] = in[r*cols + c] * scale  (f32 -> bf16 transpose, LoRA mats)
__global__ void k_transpose(const float* __restrict__ in, __bf16* __restrict__ out,
                            int rows, int cols, float scale) {
  const int idx = blockIdx.x * 256 + threadIdx.x;
  if (idx >= rows * cols) return;
  const int c = idx / rows, r = idx - c * rows;
  out[idx] = (__bf16)(in[(size_t)r * cols + c] * scale);
}

// x += g[d] * y  (LayerScale residual)
__global__ void k_residual_ls(float* __restrict__ x, const float* __restrict__ y,
                              const float* __restrict__ g, int D, int n) {
  const int idx = blockIdx.x * 256 + threadIdx.x;
  if (idx < n) x[idx] += g[idx % D] * y[idx];
}

// SwiGLU: silu(h[:, :DFF]) * h[:, DFF:]  -> bf16
__global__ void k_swiglu(const float* __restrict__ h, __bf16* __restrict__ out,
                         int S, int dff) {
  const int idx = blockIdx.x * 256 + threadIdx.x;
  if (idx >= S * dff) return;
  const int s = idx / dff, j = idx - s * dff;
  const float gt = h[(size_t)s * 2 * dff + j];
  const float ac = h[(size_t)s * 2 * dff + dff + j];
  const float sg = gt / (1.0f + __expf(-gt));
  out[idx] = (__bf16)(sg * ac);
}

// split qkv [S,3,H,HD] -> Q[H,S,HD]*(1/sqrt(HD)), K[H,S,HD], Vt[H,VROWS,SPAD]
__global__ void k_qkv_split(const float* __restrict__ qkv, __bf16* __restrict__ Q,
                            __bf16* __restrict__ Kh, __bf16* __restrict__ Vt,
                            int S, int H, int hd) {
  const int idx = blockIdx.x * 256 + threadIdx.x;
  const int D = H * hd;
  if (idx >= S * D) return;
  const int s = idx / D;
  const int t = idx - s * D;
  const int h = t / hd, d = t - h * hd;
  const size_t base = (size_t)s * 3 * D + (size_t)h * hd + d;
  Q [((size_t)h * S + s) * hd + d] = (__bf16)(qkv[base] * 0.125f); // 1/sqrt(64)
  Kh[((size_t)h * S + s) * hd + d] = (__bf16)qkv[base + D];
  Vt[((size_t)h * VROWS + d) * SPAD + s] = (__bf16)qkv[base + 2 * D];
}

// zero the padded tail columns of V^T rows 0..HD-1  (n = H*HD*16)
__global__ void k_vt_pad(__bf16* __restrict__ Vt, int S, int n) {
  const int idx = blockIdx.x * 256 + threadIdx.x;
  if (idx >= n) return;
  const int c = idx & 15;
  const int d = (idx >> 4) & (HD - 1);
  const int h = idx >> 10;
  Vt[((size_t)h * VROWS + d) * SPAD + S + c] = (__bf16)0.0f;
}

// fill V^T rows HD..HD+15: row HD = ones (rowsum tap), rest zero (n = H*16*SPAD)
__global__ void k_vt_ones(__bf16* __restrict__ Vt, int S, int n) {
  const int idx = blockIdx.x * 256 + threadIdx.x;
  if (idx >= n) return;
  const int c  = idx % SPAD;
  const int rr = (idx / SPAD) & 15;
  const int h  = idx / (SPAD * 16);
  const float v = (rr == 0 && c < S) ? 1.0f : 0.0f;
  Vt[((size_t)h * VROWS + HD + rr) * SPAD + c] = (__bf16)v;
}

// ---------------- host-side helpers ----------------
static inline int nblk(int n) { return (n + 255) / 256; }

static void gemm_f32B(const __bf16* A, int lda, const float* B, int ldb,
                      float* C, int ldc, int M, int N, int K, hipStream_t st) {
  dim3 g(N / 64, M / 128, 1);
  k_gemm_tdm<<<g, 256, 0, st>>>(A, lda, B, ldb, C, ldc, M, N, K);
}

static void gemm_bf16B(const __bf16* A, int lda, const __bf16* B, int ldb,
                       float* C, int ldc, int M, int N, int K, bool acc,
                       hipStream_t st) {
  dim3 g((N + 63) / 64, (M + 127) / 128, 1);
  if (acc) k_gemm_nt<true ><<<g, 256, 0, st>>>(A, lda, B, ldb, C, ldc, M, N, K);
  else     k_gemm_nt<false><<<g, 256, 0, st>>>(A, lda, B, ldb, C, ldc, M, N, K);
}

extern "C" void kernel_launch(void* const* d_in, const int* in_sizes, int n_in,
                              void* d_out, int out_size, void* d_ws, size_t ws_size,
                              hipStream_t stream) {
  (void)in_sizes; (void)n_in; (void)out_size; (void)ws_size;
  const int*   ids    = (const int*)  d_in[0];
  const float* emb    = (const float*)d_in[1];
  const float* ln1_g  = (const float*)d_in[2];
  const float* ln1_b  = (const float*)d_in[3];
  const float* qkv_W  = (const float*)d_in[4];
  const float* qkv_A  = (const float*)d_in[5];
  const float* qkv_B  = (const float*)d_in[6];
  const float* out_W  = (const float*)d_in[7];
  const float* out_A  = (const float*)d_in[8];
  const float* out_B  = (const float*)d_in[9];
  const float* ln2_g  = (const float*)d_in[10];
  const float* ln2_b  = (const float*)d_in[11];
  const float* fc1_W  = (const float*)d_in[12];
  const float* fc1_A  = (const float*)d_in[13];
  const float* fc1_B  = (const float*)d_in[14];
  const float* fc2_W  = (const float*)d_in[15];
  const float* fc2_A  = (const float*)d_in[16];
  const float* fc2_B  = (const float*)d_in[17];
  const float* ls1    = (const float*)d_in[18];
  const float* ls2    = (const float*)d_in[19];
  const float* lnf_g  = (const float*)d_in[20];
  const float* lnf_b  = (const float*)d_in[21];
  float* logits = (float*)d_out;

  // ---- workspace bump allocator ----
  char* wp = (char*)d_ws;
  auto alloc = [&](size_t bytes) -> void* {
    void* r = wp;
    wp += (bytes + 255) & ~(size_t)255;
    return r;
  };
  float*  x    = (float*) alloc((size_t)SS * DD * 4);
  __bf16* xnb  = (__bf16*)alloc((size_t)SS * DD * 2);
  float*  qkvf = (float*) alloc((size_t)SS * 3 * DD * 4);
  float*  tf   = (float*) alloc((size_t)SS * RR * 4);
  __bf16* tb   = (__bf16*)alloc((size_t)SS * RR * 2);
  __bf16* Qb   = (__bf16*)alloc((size_t)SS * DD * 2);
  __bf16* Kb   = (__bf16*)alloc((size_t)SS * DD * 2);
  __bf16* Vtb  = (__bf16*)alloc((size_t)HH * VROWS * SPAD * 2);
  __bf16* aob  = (__bf16*)alloc((size_t)SS * DD * 2);
  float*  yf   = (float*) alloc((size_t)SS * DD * 4);
  float*  hf   = (float*) alloc((size_t)SS * 2 * DFF * 4);
  __bf16* gb   = (__bf16*)alloc((size_t)SS * DFF * 2);
  // transposed bf16 LoRA mats (per layer)
  __bf16* qkvAT = (__bf16*)alloc((size_t)LL * RR * DD  * 2); // [R,D]
  __bf16* outAT = (__bf16*)alloc((size_t)LL * RR * DD  * 2); // [R,D]
  __bf16* fc1AT = (__bf16*)alloc((size_t)LL * RR * DD  * 2); // [R,D]
  __bf16* fc2AT = (__bf16*)alloc((size_t)LL * RR * DFF * 2); // [R,DFF]
  __bf16* qkvBT = (__bf16*)alloc((size_t)LL * 3 * DD  * RR * 2); // [3D,R] scaled
  __bf16* outBT = (__bf16*)alloc((size_t)LL * DD      * RR * 2); // [D,R]
  __bf16* fc1BT = (__bf16*)alloc((size_t)LL * 2 * DFF * RR * 2); // [2DFF,R]
  __bf16* fc2BT = (__bf16*)alloc((size_t)LL * DD      * RR * 2); // [D,R]

  // ---- pre-transpose LoRA weights to NT bf16 (scale folded into B) ----
  for (int l = 0; l < LL; ++l) {
    k_transpose<<<nblk(DD * RR), 256, 0, stream>>>(
        qkv_A + (size_t)l * DD * RR, qkvAT + (size_t)l * RR * DD, DD, RR, 1.0f);
    k_transpose<<<nblk(RR * 3 * DD), 256, 0, stream>>>(
        qkv_B + (size_t)l * RR * 3 * DD, qkvBT + (size_t)l * 3 * DD * RR, RR, 3 * DD, LORA_SCALE);
    k_transpose<<<nblk(DD * RR), 256, 0, stream>>>(
        out_A + (size_t)l * DD * RR, outAT + (size_t)l * RR * DD, DD, RR, 1.0f);
    k_transpose<<<nblk(RR * DD), 256, 0, stream>>>(
        out_B + (size_t)l * RR * DD, outBT + (size_t)l * DD * RR, RR, DD, LORA_SCALE);
    k_transpose<<<nblk(DD * RR), 256, 0, stream>>>(
        fc1_A + (size_t)l * DD * RR, fc1AT + (size_t)l * RR * DD, DD, RR, 1.0f);
    k_transpose<<<nblk(RR * 2 * DFF), 256, 0, stream>>>(
        fc1_B + (size_t)l * RR * 2 * DFF, fc1BT + (size_t)l * 2 * DFF * RR, RR, 2 * DFF, LORA_SCALE);
    k_transpose<<<nblk(DFF * RR), 256, 0, stream>>>(
        fc2_A + (size_t)l * DFF * RR, fc2AT + (size_t)l * RR * DFF, DFF, RR, 1.0f);
    k_transpose<<<nblk(RR * DD), 256, 0, stream>>>(
        fc2_B + (size_t)l * RR * DD, fc2BT + (size_t)l * DD * RR, RR, DD, LORA_SCALE);
  }

  // ---- embedding gather ----
  k_embed<<<nblk(SS * DD), 256, 0, stream>>>(ids, emb, x, SS, DD);

  // LoRA accumulate: C += ((Xb @ At^T)_bf16 @ Bt^T), At [R,K], Bt [Nout,R]
  auto lora_acc = [&](const __bf16* Xb, int ldx, int Kdim, const __bf16* At,
                      const __bf16* Bt, float* C, int Nout) {
    gemm_bf16B(Xb, ldx, At, Kdim, tf, RR, SS, RR, Kdim, false, stream);
    k_cvt<<<nblk(SS * RR), 256, 0, stream>>>(tf, tb, SS * RR);
    gemm_bf16B(tb, RR, Bt, RR, C, Nout, SS, Nout, RR, true, stream);
  };

  // ---- transformer layers ----
  for (int l = 0; l < LL; ++l) {
    const float* qW  = qkv_W + (size_t)l * 3 * DD * DD;
    const float* oW  = out_W + (size_t)l * DD * DD;
    const float* f1W = fc1_W + (size_t)l * 2 * DFF * DD;
    const float* f2W = fc2_W + (size_t)l * DD * DFF;

    // LN1 -> bf16
    k_layernorm<<<SS, 256, 0, stream>>>(x, ln1_g + (size_t)l * DD,
                                        ln1_b + (size_t)l * DD, xnb, DD);
    // qkv = xn @ W^T (+ LoRA)
    gemm_f32B(xnb, DD, qW, DD, qkvf, 3 * DD, SS, 3 * DD, DD, stream);
    lora_acc(xnb, DD, DD, qkvAT + (size_t)l * RR * DD,
             qkvBT + (size_t)l * 3 * DD * RR, qkvf, 3 * DD);

    // split -> Q (scaled), K, V^T (padded + ones row)
    k_qkv_split<<<nblk(SS * DD), 256, 0, stream>>>(qkvf, Qb, Kb, Vtb, SS, HH, HD);
    k_vt_pad<<<nblk(HH * HD * 16), 256, 0, stream>>>(Vtb, SS, HH * HD * 16);
    k_vt_ones<<<nblk(HH * 16 * SPAD), 256, 0, stream>>>(Vtb, SS, HH * 16 * SPAD);

    // fused flash attention -> aob bf16 [S, D]
    k_flash<<<dim3(SS / 64, HH), 128, 0, stream>>>(Qb, Kb, Vtb, aob, SS);

    // out proj (+ LoRA), LayerScale residual
    gemm_f32B(aob, DD, oW, DD, yf, DD, SS, DD, DD, stream);
    lora_acc(aob, DD, DD, outAT + (size_t)l * RR * DD,
             outBT + (size_t)l * DD * RR, yf, DD);
    k_residual_ls<<<nblk(SS * DD), 256, 0, stream>>>(x, yf, ls1 + (size_t)l * DD,
                                                     DD, SS * DD);

    // LN2 -> FFN (SwiGLU) -> LayerScale residual
    k_layernorm<<<SS, 256, 0, stream>>>(x, ln2_g + (size_t)l * DD,
                                        ln2_b + (size_t)l * DD, xnb, DD);
    gemm_f32B(xnb, DD, f1W, DD, hf, 2 * DFF, SS, 2 * DFF, DD, stream);
    lora_acc(xnb, DD, DD, fc1AT + (size_t)l * RR * DD,
             fc1BT + (size_t)l * 2 * DFF * RR, hf, 2 * DFF);
    k_swiglu<<<nblk(SS * DFF), 256, 0, stream>>>(hf, gb, SS, DFF);
    gemm_f32B(gb, DFF, f2W, DFF, yf, DD, SS, DD, DFF, stream);
    lora_acc(gb, DFF, DFF, fc2AT + (size_t)l * RR * DFF,
             fc2BT + (size_t)l * DD * RR, yf, DD);
    k_residual_ls<<<nblk(SS * DD), 256, 0, stream>>>(x, yf, ls2 + (size_t)l * DD,
                                                     DD, SS * DD);
  }

  // ---- final LN + tied LM head ----
  k_layernorm<<<SS, 256, 0, stream>>>(x, lnf_g, lnf_b, xnb, DD);
  gemm_f32B(xnb, DD, emb, DD, logits, VV, SS, VV, DD, stream);
}
